// MixtralSparTAMoeBlock_7164005450200
// MI455X (gfx1250) — matrix operand order repaired and checked by reading, not compiled, not run on hardware
//
#include <hip/hip_runtime.h>
#include <hip/hip_bf16.h>

// MoE (Mixtral-style) top-2 routed SwiGLU FFN for gfx1250.
// x:[T,H] f32, gate_w:[E,H], w1:[E,H,F], w3:[E,H,F], w2:[E,F,H] (all f32)
// out = concat( out[T,H] f32 , router_logits[T,E] f32 )

#define T_TOK 2048
#define H_DIM 1024
#define F_DIM 3584
#define NE 8

#define TRASH_ROW (2 * T_TOK)          // padding row of h-pool for invalid tile slots
#define ROWS_POOL (2 * T_TOK + 16)

#define BM 64                           // gathered token rows per block (4 sub-tiles of 16)
#define KC 256                          // K chunk staged in LDS
#define LDS_PAD 264                     // 256 + 8 bf16 -> 4-bank step/row, 16B-aligned rows

typedef __attribute__((ext_vector_type(16))) __bf16 v16bf;
typedef __attribute__((ext_vector_type(8)))  __bf16 v8bf;
typedef __attribute__((ext_vector_type(4)))  __bf16 v4bf;
typedef __attribute__((ext_vector_type(8)))  float  v8f;
typedef __attribute__((ext_vector_type(4)))  float  v4f;

// ---------------------------------------------------------------------------
// WMMA fragment helpers (layouts per CDNA5 ISA 7.12.2, wave32)
// ---------------------------------------------------------------------------

// A fragment, 16x32 bf16, from an LDS tile (row-major, stride lda elems).
// lanes 0-15: M=lane, K in {0..7, 16..23}; lanes 16-31: M=lane-16, K in {8..15, 24..31}
// Elements are two contiguous 16B runs -> two ds_load_b128 (alignment guaranteed:
// lda=264 -> 528B row stride; k0 % 32 == 0; base in {0,8}).
__device__ inline v16bf load_frag_a_lds(const __bf16* lds, int lda, int k0, int lane) {
    int m    = lane & 15;
    int base = (lane >> 4) * 8;
    const __bf16* row = lds + m * lda + k0 + base;
    union { struct { v8bf lo, hi; } s; v16bf v; } u;
    u.s.lo = *(const v8bf*)(row);        // k = base + 0..7
    u.s.hi = *(const v8bf*)(row + 16);   // k = 16 + base + 0..7
    return u.v;
}

// B fragment, 32x16 bf16 from a global f32 K-major matrix (row stride ldb elems).
// lanes 0-15: N=lane, K in {0..7, 16..23}; lanes 16-31: N=lane-16, K in {8..15, 24..31}
__device__ inline v16bf load_frag_b_global(const float* __restrict__ B, long ldb,
                                           int k0, int ncol, int lane) {
    int n    = lane & 15;
    int koff = (lane >> 4) * 8;
    const float* p = B + (long)(k0 + koff) * ldb + ncol + n;
    v16bf b;
#pragma unroll
    for (int j = 0; j < 16; ++j) b[j] = (__bf16)p[(long)(j + (j & 8)) * ldb];
    return b;
}

// ---------------------------------------------------------------------------
// Kernel 0: zero the output accumulator and the per-expert counters
// ---------------------------------------------------------------------------
__global__ void zero_init_kernel(float* __restrict__ out, int n, int* __restrict__ counts) {
    int i = blockIdx.x * blockDim.x + threadIdx.x;
    if (i < n) out[i] = 0.f;
    if (blockIdx.x == 0 && threadIdx.x < NE) counts[threadIdx.x] = 0;
}

// ---------------------------------------------------------------------------
// Kernel 1: router — logits, softmax, top-2, renorm, expert token gather lists
// One wave32 per token.
// ---------------------------------------------------------------------------
__global__ __launch_bounds__(256) void router_kernel(
    const float* __restrict__ x, const float* __restrict__ gw,
    float* __restrict__ logits_out, int* __restrict__ counts,
    int* __restrict__ tokenlist, int* __restrict__ rowlist, float* __restrict__ wlist)
{
    int lane = threadIdx.x & 31;
    int wave = threadIdx.x >> 5;
    int t = blockIdx.x * 8 + wave;
    if (t >= T_TOK) return;

    float acc[NE];
#pragma unroll
    for (int e = 0; e < NE; ++e) acc[e] = 0.f;

    const float* xp = x + (size_t)t * H_DIM;
    for (int h = lane; h < H_DIM; h += 32) {
        float xv = xp[h];
#pragma unroll
        for (int e = 0; e < NE; ++e) acc[e] += xv * gw[e * H_DIM + h];
    }
#pragma unroll
    for (int e = 0; e < NE; ++e) {
#pragma unroll
        for (int off = 16; off > 0; off >>= 1)
            acc[e] += __shfl_xor(acc[e], off, 32);
    }

    if (lane == 0) {
        float mx = acc[0];
#pragma unroll
        for (int e = 1; e < NE; ++e) mx = fmaxf(mx, acc[e]);
        float ex[NE];
#pragma unroll
        for (int e = 0; e < NE; ++e) {
            ex[e] = __expf(acc[e] - mx);
            logits_out[t * NE + e] = acc[e];      // raw logits (second output)
        }
        // top-2 (softmax denominator cancels in the renormalization)
        int i0 = 0; float p0 = ex[0];
#pragma unroll
        for (int e = 1; e < NE; ++e) if (ex[e] > p0) { p0 = ex[e]; i0 = e; }
        int i1 = (i0 == 0) ? 1 : 0; float p1 = ex[i1];
#pragma unroll
        for (int e = 0; e < NE; ++e) if (e != i0 && ex[e] > p1) { p1 = ex[e]; i1 = e; }
        float inv = 1.f / (p0 + p1);

        int pos0 = atomicAdd(&counts[i0], 1);
        tokenlist[i0 * T_TOK + pos0] = t;
        rowlist  [i0 * T_TOK + pos0] = 2 * t;
        wlist    [i0 * T_TOK + pos0] = p0 * inv;

        int pos1 = atomicAdd(&counts[i1], 1);
        tokenlist[i1 * T_TOK + pos1] = t;
        rowlist  [i1 * T_TOK + pos1] = 2 * t + 1;
        wlist    [i1 * T_TOK + pos1] = p1 * inv;
    }
}

// ---------------------------------------------------------------------------
// Kernel 2: up-projection h = silu(x@w1) * (x@w3), gathered per expert.
// Block: 64 gathered tokens x 128 F columns. 8 waves, each one 16-col strip,
// four 16-row A fragments, B (w1,w3) reused across all -> 8 WMMAs per K step.
// ---------------------------------------------------------------------------
__global__ __launch_bounds__(256) void moe_ffn_up(
    const float* __restrict__ x, const float* __restrict__ w1, const float* __restrict__ w3,
    const int* __restrict__ counts, const int* __restrict__ tokenlist,
    const int* __restrict__ rowlist, __bf16* __restrict__ hpool)
{
    int e   = blockIdx.x >> 5;
    int rt  = blockIdx.x & 31;
    int cnt = counts[e];
    if (rt * BM >= cnt) return;                    // block-uniform: EXEC stays full for WMMA
    int fbase = blockIdx.y * 128;
    int lane  = threadIdx.x & 31;
    int wave  = threadIdx.x >> 5;

    __shared__ __align__(16) __bf16 ldsA[BM * LDS_PAD];
    __shared__ int t_sh[BM];
    __shared__ int row_sh[BM];

    if (threadIdx.x < BM) {
        int pos = rt * BM + (int)threadIdx.x;
        bool valid = pos < cnt;
        int pc = valid ? pos : (cnt - 1);
        t_sh[threadIdx.x]   = tokenlist[e * T_TOK + pc];
        row_sh[threadIdx.x] = valid ? rowlist[e * T_TOK + pos] : TRASH_ROW;
    }
    __syncthreads();

    const float* w1e = w1 + (size_t)e * H_DIM * F_DIM;
    const float* w3e = w3 + (size_t)e * H_DIM * F_DIM;
    int fcol = fbase + wave * 16;

    v8f acc1[4] = {};
    v8f acc3[4] = {};

    for (int kc = 0; kc < H_DIM; kc += KC) {
        __syncthreads();
        // stage x chunk [BM rows x KC] as bf16; float4 loads, 8B LDS stores
        for (int idx = threadIdx.x; idx < BM * (KC / 4); idx += 256) {
            int m  = idx >> 6;                 // KC/4 == 64 quads per row
            int k4 = (idx & 63) * 4;
            v4f xv = *(const v4f*)(x + (size_t)t_sh[m] * H_DIM + kc + k4);
            v4bf bb;
#pragma unroll
            for (int i = 0; i < 4; ++i) bb[i] = (__bf16)xv[i];
            *(v4bf*)&ldsA[m * LDS_PAD + k4] = bb;
        }
        __syncthreads();

        // prefetch next weight chunk (global_prefetch_b8)
        if (kc + KC < H_DIM) {
            __builtin_prefetch(w1e + (size_t)(kc + KC) * F_DIM + fcol + lane, 0, 0);
            __builtin_prefetch(w3e + (size_t)(kc + KC) * F_DIM + fcol + lane, 0, 0);
        }

#pragma unroll
        for (int k0 = 0; k0 < KC; k0 += 32) {
            v16bf b1 = load_frag_b_global(w1e, F_DIM, kc + k0, fcol, lane);
            v16bf b3 = load_frag_b_global(w3e, F_DIM, kc + k0, fcol, lane);
#pragma unroll
            for (int s = 0; s < 4; ++s) {
                v16bf a = load_frag_a_lds(ldsA + s * 16 * LDS_PAD, LDS_PAD, k0, lane);
                acc1[s] = __builtin_amdgcn_wmma_f32_16x16x32_bf16(false, a, false, b1, (short)0, acc1[s], false, false);
                acc3[s] = __builtin_amdgcn_wmma_f32_16x16x32_bf16(false, a, false, b3, (short)0, acc3[s], false, false);
            }
        }
    }

    // SwiGLU epilogue -> bf16 h-pool (fast rcp: silu precision is ample)
    int n    = lane & 15;
    int half = lane >> 4;
#pragma unroll
    for (int s = 0; s < 4; ++s) {
#pragma unroll
        for (int r = 0; r < 8; ++r) {
            int m = s * 16 + r + 8 * half;
            float s1 = acc1[s][r];
            float sig = __builtin_amdgcn_rcpf(1.f + __expf(-s1));
            float hv = s1 * sig * acc3[s][r];
            hpool[(size_t)row_sh[m] * F_DIM + fcol + n] = (__bf16)hv;
        }
    }
}

// ---------------------------------------------------------------------------
// Kernel 3: down-projection out += weight * (h @ w2), gathered per expert.
// Block: 64 h-rows x 128 H columns; h streamed through LDS in 256-wide K chunks.
// B fragment reused across all four row sub-tiles -> 4 WMMAs per K step per wave.
// ---------------------------------------------------------------------------
__global__ __launch_bounds__(256) void moe_ffn_down(
    const __bf16* __restrict__ hpool, const float* __restrict__ w2,
    const int* __restrict__ counts, const int* __restrict__ tokenlist,
    const int* __restrict__ rowlist, const float* __restrict__ wlist,
    float* __restrict__ out)
{
    int e   = blockIdx.x >> 5;
    int rt  = blockIdx.x & 31;
    int cnt = counts[e];
    if (rt * BM >= cnt) return;
    int hbase = blockIdx.y * 128;
    int lane  = threadIdx.x & 31;
    int wave  = threadIdx.x >> 5;

    __shared__ __align__(16) __bf16 ldsH[BM * LDS_PAD];
    __shared__ int   t_sh[BM];
    __shared__ int   row_sh[BM];
    __shared__ float w_sh[BM];

    if (threadIdx.x < BM) {
        int pos = rt * BM + (int)threadIdx.x;
        bool valid = pos < cnt;
        t_sh[threadIdx.x]   = valid ? tokenlist[e * T_TOK + pos] : 0;
        row_sh[threadIdx.x] = valid ? rowlist[e * T_TOK + pos] : TRASH_ROW;
        w_sh[threadIdx.x]   = valid ? wlist[e * T_TOK + pos] : 0.f;
    }
    __syncthreads();

    const float* w2e = w2 + (size_t)e * F_DIM * H_DIM;
    int hcol = hbase + wave * 16;
    v8f acc[4] = {};

    for (int kc = 0; kc < F_DIM; kc += KC) {
        __syncthreads();
        // stage h chunk [BM rows x KC] bf16; 16B loads / 16B LDS stores
        for (int idx = threadIdx.x; idx < BM * (KC / 8); idx += 256) {
            int m  = idx >> 5;                 // KC/8 == 32 octets per row
            int k8 = (idx & 31) * 8;
            v8bf hv = *(const v8bf*)(hpool + (size_t)row_sh[m] * F_DIM + kc + k8);
            *(v8bf*)&ldsH[m * LDS_PAD + k8] = hv;
        }
        __syncthreads();

        if (kc + KC < F_DIM) {
            __builtin_prefetch(w2e + (size_t)(kc + KC) * H_DIM + hcol + lane, 0, 0);
        }

#pragma unroll
        for (int k0 = 0; k0 < KC; k0 += 32) {
            v16bf b = load_frag_b_global(w2e, H_DIM, kc + k0, hcol, lane);
#pragma unroll
            for (int s = 0; s < 4; ++s) {
                v16bf a = load_frag_a_lds(ldsH + s * 16 * LDS_PAD, LDS_PAD, k0, lane);
                acc[s] = __builtin_amdgcn_wmma_f32_16x16x32_bf16(false, a, false, b, (short)0, acc[s], false, false);
            }
        }
    }

    int n    = lane & 15;
    int half = lane >> 4;
#pragma unroll
    for (int s = 0; s < 4; ++s) {
#pragma unroll
        for (int r = 0; r < 8; ++r) {
            int m   = s * 16 + r + 8 * half;
            int pos = rt * BM + m;
            if (pos < cnt)
                atomicAdd(&out[(size_t)t_sh[m] * H_DIM + hcol + n], acc[s][r] * w_sh[m]);
        }
    }
}

// ---------------------------------------------------------------------------
extern "C" void kernel_launch(void* const* d_in, const int* in_sizes, int n_in,
                              void* d_out, int out_size, void* d_ws, size_t ws_size,
                              hipStream_t stream) {
    // setup_inputs() dict order: hidden_states, gate_w, w1, w3, w2
    const float* x  = (const float*)d_in[0];
    const float* gw = (const float*)d_in[1];
    const float* w1 = (const float*)d_in[2];
    const float* w3 = (const float*)d_in[3];
    const float* w2 = (const float*)d_in[4];

    float* out        = (float*)d_out;                  // [T, H]
    float* logits_out = out + (size_t)T_TOK * H_DIM;    // [T, E]

    // workspace layout (~30 MB)
    char* ws = (char*)d_ws;
    int*    counts    = (int*)ws;                                   // NE ints
    int*    tokenlist = (int*)(ws + 256);                           // NE*T ints
    int*    rowlist   = (int*)(ws + 256 + (size_t)NE * T_TOK * 4);  // NE*T ints
    float*  wlist     = (float*)(ws + 256 + (size_t)2 * NE * T_TOK * 4);
    __bf16* hpool     = (__bf16*)(ws + 256 + (size_t)3 * NE * T_TOK * 4); // ROWS_POOL*F bf16

    int n = T_TOK * H_DIM;
    zero_init_kernel<<<(n + 255) / 256, 256, 0, stream>>>(out, n, counts);

    router_kernel<<<T_TOK / 8, 256, 0, stream>>>(x, gw, logits_out, counts,
                                                 tokenlist, rowlist, wlist);

    dim3 gA(NE * 32, F_DIM / 128);    // (256, 28); inactive tiles exit early
    moe_ffn_up<<<gA, 256, 0, stream>>>(x, w1, w3, counts, tokenlist, rowlist, hpool);

    dim3 gB(NE * 32, H_DIM / 128);    // (256, 8)
    moe_ffn_down<<<gB, 256, 0, stream>>>(hpool, w2, counts, tokenlist, rowlist, wlist, out);
}
